// MYAttention_1262720385460
// MI455X (gfx1250) — matrix operand order repaired and checked by reading, not compiled, not run on hardware
//
#include <hip/hip_runtime.h>
#include <hip/hip_bf16.h>
#include <stdint.h>

typedef __attribute__((ext_vector_type(16))) __bf16 v16bf;
typedef __attribute__((ext_vector_type(8)))  float  v8f;
typedef __attribute__((ext_vector_type(4))) unsigned int u32x4;
typedef __attribute__((ext_vector_type(8))) int          i32x8;
typedef __attribute__((ext_vector_type(4))) int          i32x4;

#define B_SZ 64
#define N_SZ 1024
#define F_SZ 1024
#define H_SZ 1024
#define A_SZ 512
#define NCH  (F_SZ / 64)          // 16 K-chunks of 64

#define GLB_AS __attribute__((address_space(1)))
#define LDS_AS __attribute__((address_space(3)))

// ---- feature detection for the CDNA5 data-mover paths ----------------------
#if defined(__has_builtin)
# if __has_builtin(__builtin_amdgcn_tensor_load_to_lds) && \
     __has_builtin(__builtin_amdgcn_s_wait_tensorcnt)
#  define USE_TDM 1
#  if __has_include(<hip/amd_detail/amd_gfx1250_TDM.h>)
#   define TDM_6ARG 1             // clang-23 / therock headers: 6-arg builtin
#  endif
# endif
# if __has_builtin(__builtin_amdgcn_global_load_async_to_lds_b128) && \
     __has_builtin(__builtin_amdgcn_s_wait_asynccnt)
#  define HAVE_ASYNC 1
# endif
#endif

__device__ __forceinline__ unsigned short f2bf(float x) {
    unsigned u = __float_as_uint(x);
    u += 0x7FFFu + ((u >> 16) & 1u);          // round-to-nearest-even
    return (unsigned short)(u >> 16);
}

// ---------------------------------------------------------------------------
// Kernel 1: Wv (F x A, fp32) -> wvt (A x F, bf16), tiled transpose via LDS.
// 1MB bf16 result stays L2-resident (192MB L2) for the whole GEMM.
// ---------------------------------------------------------------------------
__global__ void k_wvt(const float* __restrict__ Wv, unsigned short* __restrict__ wvt) {
    __shared__ float tile[32][33];
    const int kt = blockIdx.x * 32;   // F dimension
    const int at = blockIdx.y * 32;   // A dimension
    const int tx = threadIdx.x, ty = threadIdx.y; // block (32, 8)
    #pragma unroll
    for (int i = 0; i < 32; i += 8)
        tile[ty + i][tx] = Wv[(size_t)(kt + ty + i) * A_SZ + at + tx];
    __syncthreads();
    #pragma unroll
    for (int i = 0; i < 32; i += 8)
        wvt[(size_t)(at + ty + i) * F_SZ + kt + tx] = f2bf(tile[tx][ty + i]);
}

// ---------------------------------------------------------------------------
// Kernel 2: hb[b][a] = key[b] @ Wh[:,a] + bh[a] + bv[a]   (fold bv in here)
// ---------------------------------------------------------------------------
__global__ void k_h(const float* __restrict__ key, const float* __restrict__ Wh,
                    const float* __restrict__ bh, const float* __restrict__ bv,
                    float* __restrict__ hb) {
    const int b = blockIdx.x >> 1;
    const int a = ((blockIdx.x & 1) << 8) + threadIdx.x;
    float acc = bh[a] + bv[a];
    const float* kp = key + (size_t)b * H_SZ;
    for (int k = 0; k < H_SZ; ++k)
        acc += kp[k] * Wh[(size_t)k * A_SZ + a];
    hb[(size_t)b * A_SZ + a] = acc;
}

// ---------------------------------------------------------------------------
// TDM descriptor issue: 2D tile, 64 rows x 64 fp32, row stride 1024 floats.
// D# group0: count=1 | lds_addr | global_addr | type=2 ("image")
// D# group1: data_size=4B, tensor_dim0=1024, tensor_dim1=1024,
//            tile_dim0=64, tile_dim1=64, tensor_dim0_stride=1024
// ---------------------------------------------------------------------------
#if defined(USE_TDM)
__device__ __forceinline__ void tdm_issue(const float* gsrc, unsigned lds_byte_addr) {
    unsigned long long ga = (unsigned long long)(uintptr_t)gsrc;
    u32x4 g0;
    g0[0] = 1u;                                             // count=1, user desc
    g0[1] = lds_byte_addr;                                  // lds_addr [63:32]
    g0[2] = (unsigned)ga;                                   // global_addr lo
    g0[3] = (unsigned)((ga >> 32) & 0x01FFFFFFu) | 0x80000000u; // ga hi | type=2
    i32x8 g1;
    g1[0] = 0x00020000;                 // wg_mask=0, data_size=2 (4 bytes)
    g1[1] = (int)(1024u << 16);         // tensor_dim0[15:0] = 1024 @ bits 63:48
    g1[2] = (int)(1024u << 16);         // dim0 hi = 0 ; tensor_dim1[15:0] = 1024
    g1[3] = (int)(64u << 16);           // dim1 hi = 0 ; tile_dim0 = 64
    g1[4] = 64;                         // tile_dim1 = 64, tile_dim2 = 0
    g1[5] = 1024;                       // tensor_dim0_stride = 1024 elements
    g1[6] = 0;                          // stride hi / tensor_dim1_stride lo
    g1[7] = 0;
    i32x4 z4 = {0, 0, 0, 0};            // groups 2/3 unused for 2D tile
#if defined(TDM_6ARG)
    i32x8 z8 = {0, 0, 0, 0, 0, 0, 0, 0};
    __builtin_amdgcn_tensor_load_to_lds(g0, g1, z4, z4, z8, 0);
#else
    __builtin_amdgcn_tensor_load_to_lds(g0, g1, z4, z4, 0);
#endif
}
#endif

// ---------------------------------------------------------------------------
// Kernel 3 (the big one): fused  scores[b][n] = wa . tanh(hb + feats@Wv)
// Workgroup: 8 waves = 256 threads, computes 64 seq rows x 512 A cols.
// Wave w: 4 M-tiles x 4 N-tiles, K = 1024 via V_WMMA_F32_16X16X32_BF16.
// feats chunk staged fp32 -> LDS by TDM (double-buffered, TENSORcnt) or
// async-copy (ASYNCcnt) or direct loads, then converted to bf16 in the
// WMMA-A-friendly layout.  Wv read as bf16 [A][F] (K-contiguous) from L2.
// ---------------------------------------------------------------------------
__global__ void __launch_bounds__(256) k_score(
    const float* __restrict__ feats, const unsigned short* __restrict__ wvt,
    const float* __restrict__ hb, const float* __restrict__ wa,
    float* __restrict__ scores)
{
#if defined(USE_TDM)
    __shared__ __align__(16) float sF[2][64 * 64];       // raw fp32, 2 x 16KB
#elif defined(HAVE_ASYNC)
    __shared__ __align__(16) float sF[1][64 * 64];       // raw fp32, 16KB
#endif
    __shared__ __align__(16) unsigned short sA[64 * 64]; // [row][k] bf16, 8KB
    __shared__ float sScore[64];

    const int b     = blockIdx.x >> 4;
    const int row0  = (blockIdx.x & 15) << 6;
    const int tid   = threadIdx.x;
    const int wave  = tid >> 5;
    const int lane  = tid & 31;
    const int lhalf = lane >> 4;
    const int l16   = lane & 15;
    const int colbase = wave << 6;              // 64 A-columns per wave

    if (tid < 64) sScore[tid] = 0.f;

    v8f acc[4][4];
    #pragma unroll
    for (int mt = 0; mt < 4; ++mt)
        #pragma unroll
        for (int nt = 0; nt < 4; ++nt)
            #pragma unroll
            for (int e = 0; e < 8; ++e) acc[mt][nt][e] = 0.f;

    const float* fb = feats + ((size_t)b * N_SZ + row0) * F_SZ;

#if defined(USE_TDM)
    const unsigned ldsF0 = (unsigned)(uintptr_t)(LDS_AS char*)&sF[0][0];
    const unsigned ldsF1 = (unsigned)(uintptr_t)(LDS_AS char*)&sF[1][0];
    if (wave == 0) tdm_issue(fb, ldsF0);        // prologue: DMA chunk 0
#endif

    for (int c = 0; c < NCH; ++c) {
#if defined(USE_TDM)
        // ---- TDM double-buffered staging: DMA c+1 overlaps compute of c ----
        if (wave == 0) {
            if (c + 1 < NCH) {
                tdm_issue(fb + (c + 1) * 64, ((c + 1) & 1) ? ldsF1 : ldsF0);
                __builtin_amdgcn_s_wait_tensorcnt(1);   // chunk c landed
            } else {
                __builtin_amdgcn_s_wait_tensorcnt(0);
            }
        }
        __syncthreads();
        #pragma unroll
        for (int i = 0; i < 4; ++i) {           // LDS fp32 -> LDS bf16
            const int idx = i * 256 + tid;
            const int row = idx >> 4;
            const int kq  = (idx & 15) << 2;
            const float4 v4 = *(const float4*)&sF[c & 1][row * 64 + kq];
            unsigned short* d = &sA[row * 64 + kq];
            d[0] = f2bf(v4.x); d[1] = f2bf(v4.y); d[2] = f2bf(v4.z); d[3] = f2bf(v4.w);
        }
        __syncthreads();
#elif defined(HAVE_ASYNC)
        // ---- ASYNCcnt-tracked copy: each thread copies what it converts ----
        #pragma unroll
        for (int i = 0; i < 4; ++i) {
            const int idx = i * 256 + tid;
            const int row = idx >> 4;
            const int kq  = (idx & 15) << 2;
            __builtin_amdgcn_global_load_async_to_lds_b128(
                (GLB_AS i32x4*)(fb + (size_t)row * F_SZ + c * 64 + kq),
                (LDS_AS i32x4*)&sF[0][row * 64 + kq], 0, 0);
        }
        __builtin_amdgcn_s_wait_asynccnt(0);
        __syncthreads();
        #pragma unroll
        for (int i = 0; i < 4; ++i) {
            const int idx = i * 256 + tid;
            const int row = idx >> 4;
            const int kq  = (idx & 15) << 2;
            const float4 v4 = *(const float4*)&sF[0][row * 64 + kq];
            unsigned short* d = &sA[row * 64 + kq];
            d[0] = f2bf(v4.x); d[1] = f2bf(v4.y); d[2] = f2bf(v4.z); d[3] = f2bf(v4.w);
        }
        __syncthreads();
#else
        // ---- direct-load fallback ----
        __syncthreads();
        #pragma unroll
        for (int i = 0; i < 4; ++i) {
            const int idx = i * 256 + tid;
            const int row = idx >> 4;
            const int kq  = (idx & 15) << 2;
            const float4 v4 = *(const float4*)(fb + (size_t)row * F_SZ + c * 64 + kq);
            unsigned short* d = &sA[row * 64 + kq];
            d[0] = f2bf(v4.x); d[1] = f2bf(v4.y); d[2] = f2bf(v4.z); d[3] = f2bf(v4.w);
        }
        __syncthreads();
#endif
        if (c + 1 < NCH)                         // hint next feats chunk
            __builtin_prefetch(fb + (size_t)(tid >> 2) * F_SZ + (c + 1) * 64, 0, 1);

        #pragma unroll
        for (int ks = 0; ks < 2; ++ks) {         // two K=32 steps per chunk
            const int bk = ks * 32;
            // A fragments (16x32 bf16): M=lane%16, K=(lane/16)*16 + e
            v16bf afrag[4];
            #pragma unroll
            for (int mt = 0; mt < 4; ++mt) {
                const int m = mt * 16 + l16;
                union { uint4 q[2]; v16bf v; } u;
                u.q[0] = *(const uint4*)&sA[m * 64 + bk + lhalf * 16];
                u.q[1] = *(const uint4*)&sA[m * 64 + bk + lhalf * 16 + 8];
                afrag[mt] = u.v;
            }
            const int kg = c * 64 + bk + lhalf * 16;
            #pragma unroll
            for (int nt = 0; nt < 4; ++nt) {
                // B fragment (32x16 bf16): N=lane%16, K=(lane/16)*16 + e
                const int a = colbase + nt * 16 + l16;
                union { uint4 q[2]; v16bf v; } w;
                const uint4* p = (const uint4*)&wvt[(size_t)a * F_SZ + kg];
                w.q[0] = p[0]; w.q[1] = p[1];
                #pragma unroll
                for (int mt = 0; mt < 4; ++mt) {
                    acc[mt][nt] = __builtin_amdgcn_wmma_f32_16x16x32_bf16(
                        false, afrag[mt], false, w.v, (short)0, acc[mt][nt],
                        false, false);
                }
            }
        }
    }
    __syncthreads();

    // ---- fused epilogue: tanh(hb + v) . wa, reduce over A into sScore ----
    float hwv[4], wav[4];
    #pragma unroll
    for (int nt = 0; nt < 4; ++nt) {
        const int a = colbase + nt * 16 + l16;
        hwv[nt] = hb[(size_t)b * A_SZ + a];
        wav[nt] = wa[a];
    }
    #pragma unroll
    for (int mt = 0; mt < 4; ++mt) {
        #pragma unroll
        for (int r = 0; r < 8; ++r) {            // C/D layout: M = mt*16 + lhalf*8 + r
            float part = 0.f;
            #pragma unroll
            for (int nt = 0; nt < 4; ++nt)
                part += tanhf(hwv[nt] + acc[mt][nt][r]) * wav[nt];
            atomicAdd(&sScore[mt * 16 + lhalf * 8 + r], part);  // ds_add_f32
        }
    }
    __syncthreads();
    if (tid < 64) scores[(size_t)b * N_SZ + row0 + tid] = sScore[tid];
}

// ---------------------------------------------------------------------------
// Kernel 4: softmax over N per batch -> alpha
// ---------------------------------------------------------------------------
__global__ void k_softmax(const float* __restrict__ scores, float* __restrict__ alpha) {
    __shared__ float red[256];
    const int b = blockIdx.x, tid = threadIdx.x;
    const float* s = scores + (size_t)b * N_SZ;
    float v[4], m = -1e30f;
    #pragma unroll
    for (int i = 0; i < 4; ++i) { v[i] = s[i * 256 + tid]; m = fmaxf(m, v[i]); }
    red[tid] = m; __syncthreads();
    for (int o = 128; o > 0; o >>= 1) {
        if (tid < o) red[tid] = fmaxf(red[tid], red[tid + o]);
        __syncthreads();
    }
    m = red[0]; __syncthreads();
    float e[4], sum = 0.f;
    #pragma unroll
    for (int i = 0; i < 4; ++i) { e[i] = __expf(v[i] - m); sum += e[i]; }
    red[tid] = sum; __syncthreads();
    for (int o = 128; o > 0; o >>= 1) {
        if (tid < o) red[tid] += red[tid + o];
        __syncthreads();
    }
    const float inv = 1.f / red[0];
    #pragma unroll
    for (int i = 0; i < 4; ++i)
        alpha[(size_t)b * N_SZ + i * 256 + tid] = e[i] * inv;
}

// ---------------------------------------------------------------------------
// Kernel 5: att_feats[b][f] = sum_n alpha[b][n] * feats[b][n][f]  (BW-bound)
// alpha preloaded into LDS via the async copy path when available.
// ---------------------------------------------------------------------------
__global__ void k_att(const float* __restrict__ feats, const float* __restrict__ alpha,
                      float* __restrict__ out) {
    __shared__ __align__(16) float sal[N_SZ];
    const int b  = blockIdx.x >> 2;
    const int f0 = (blockIdx.x & 3) << 8;
    const int tid = threadIdx.x;
#if defined(HAVE_ASYNC)
    // 1024 floats = 256 threads x one b128 async transfer each
    __builtin_amdgcn_global_load_async_to_lds_b128(
        (GLB_AS i32x4*)(alpha + (size_t)b * N_SZ + tid * 4),
        (LDS_AS i32x4*)&sal[tid * 4], 0, 0);
    __builtin_amdgcn_s_wait_asynccnt(0);
    __syncthreads();
#else
    for (int i = tid; i < N_SZ; i += 256) sal[i] = alpha[(size_t)b * N_SZ + i];
    __syncthreads();
#endif
    const float* fp = feats + (size_t)b * N_SZ * F_SZ + f0 + tid;
    float acc = 0.f;
    for (int n = 0; n < N_SZ; ++n) acc += sal[n] * fp[(size_t)n * F_SZ];
    out[(size_t)b * F_SZ + f0 + tid] = acc;
}

// ---------------------------------------------------------------------------
extern "C" void kernel_launch(void* const* d_in, const int* in_sizes, int n_in,
                              void* d_out, int out_size, void* d_ws, size_t ws_size,
                              hipStream_t stream) {
    const float* feats = (const float*)d_in[0];
    // d_in[1] = cluster_feats : unused by the reference
    const float* key = (const float*)d_in[2];
    const float* Wv  = (const float*)d_in[3];
    const float* bv  = (const float*)d_in[4];
    const float* Wh  = (const float*)d_in[5];
    const float* bh  = (const float*)d_in[6];
    const float* wa  = (const float*)d_in[7];

    float* out_att   = (float*)d_out;                 // (B, F)
    float* out_alpha = out_att + (size_t)B_SZ * F_SZ; // (B, N)

    char* ws = (char*)d_ws;
    unsigned short* wvt = (unsigned short*)ws;                              // 1 MB
    float* hb     = (float*)(ws + (size_t)A_SZ * F_SZ * 2);                 // 128 KB
    float* scores = (float*)(ws + (size_t)A_SZ * F_SZ * 2
                                + (size_t)B_SZ * A_SZ * 4);                 // 256 KB

    k_wvt    <<<dim3(F_SZ / 32, A_SZ / 32), dim3(32, 8), 0, stream>>>(Wv, wvt);
    k_h      <<<B_SZ * 2, 256, 0, stream>>>(key, Wh, bh, bv, hb);
    k_score  <<<B_SZ * (N_SZ / 64), 256, 0, stream>>>(feats, wvt, hb, wa, scores);
    k_softmax<<<B_SZ, 256, 0, stream>>>(scores, out_alpha);
    k_att    <<<B_SZ * 4, 256, 0, stream>>>(feats, out_alpha, out_att);
}
